// PointTransformerAutoencoder_17214228922885
// MI455X (gfx1250) — matrix operand order repaired and checked by reading, not compiled, not run on hardware
//
#include <hip/hip_runtime.h>

// Problem constants (from reference): B=2, S=8, N=1024, D=3, H=64, K=16, LAT=32, NLAYERS=3
#define BSN   16384      // B*S*N rows
#define NPTS  1024       // points per (b,s)
#define HDIM  64
#define KNN   16

typedef __attribute__((ext_vector_type(16))) _Float16 v16h;
typedef __attribute__((ext_vector_type(8)))  _Float16 v8h;   // 16-byte LDS chunk
typedef __attribute__((ext_vector_type(8)))  float    v8f;

union fragAB { v16h v; v8h h2[2]; };
union fragC  { v8f  v; float f[8]; };

__device__ __forceinline__ float gelu_f(float x) {
  float x3 = x * x * x;
  return 0.5f * x * (1.f + tanhf(0.79788456080286535588f * (x + 0.044715f * x3)));
}

// ---------------------------------------------------------------------------
// x[M,64] = in3[M,3] @ W[3,64] + b   (used for enc_in and per-layer pe)
// ---------------------------------------------------------------------------
__global__ __launch_bounds__(256) void dense3_k(
    const float* __restrict__ in3, const float* __restrict__ W,
    const float* __restrict__ b, float* __restrict__ out, int M) {
  int i = blockIdx.x * blockDim.x + threadIdx.x;
  if (i >= M * HDIM) return;
  int r = i >> 6, h = i & 63;
  float p0 = in3[r * 3 + 0], p1 = in3[r * 3 + 1], p2 = in3[r * 3 + 2];
  out[i] = b[h] + p0 * W[h] + p1 * W[HDIM + h] + p2 * W[2 * HDIM + h];
}

// ---------------------------------------------------------------------------
// Brute-force KNN (top-16 smallest squared distance, includes self).
// ---------------------------------------------------------------------------
__global__ __launch_bounds__(256) void knn_k(const float* __restrict__ pts,
                                             int* __restrict__ nidx) {
  int p = blockIdx.x * blockDim.x + threadIdx.x;
  if (p >= BSN) return;
  int bs = p >> 10, n = p & 1023;
  const float* base = pts + (size_t)bs * NPTS * 3;
  float px = base[n * 3], py = base[n * 3 + 1], pz = base[n * 3 + 2];
  float bd[KNN];
  int   bi[KNN];
#pragma unroll
  for (int j = 0; j < KNN; ++j) { bd[j] = 3.4e38f; bi[j] = 0; }
  for (int m = 0; m < NPTS; ++m) {
    float dx = base[m * 3] - px, dy = base[m * 3 + 1] - py, dz = base[m * 3 + 2] - pz;
    float d2 = dx * dx + dy * dy + dz * dz;
    if (d2 < bd[KNN - 1]) {
      bd[KNN - 1] = d2; bi[KNN - 1] = m;
#pragma unroll
      for (int j = KNN - 1; j > 0; --j) {
        if (bd[j] < bd[j - 1]) {
          float td = bd[j]; bd[j] = bd[j - 1]; bd[j - 1] = td;
          int   ti = bi[j]; bi[j] = bi[j - 1]; bi[j - 1] = ti;
        }
      }
    }
  }
#pragma unroll
  for (int j = 0; j < KNN; ++j) nidx[p * KNN + j] = bi[j];
}

// ---------------------------------------------------------------------------
// WMMA fragment helpers (v_wmma_f32_16x16x32_f16, wave32).
// A 16x32 layout: lane half hh, element e<8 -> K=8*hh+e ; e>=8 -> 16+8*hh+(e-8)
//   => per lane: halfs [8*hh,+8) and [16+8*hh,+8): two b128 LDS loads.
// B 32x16 layout with W stored TRANSPOSED in LDS (wT[n][k]):
//   lane column n, same K order as A => halfs [8*hh,+8) and [16+8*hh,+8): two b128 LDS loads.
// C/D: vgpr r -> row r+8*hh, col = lane&15.
// ---------------------------------------------------------------------------
__device__ __forceinline__ void load_fragA(const _Float16* arow, int hh,
                                           fragAB& a0, fragAB& a1) {
  const v8h* ar = (const v8h*)arow;           // 8 halfs per chunk
  a0.h2[0] = ar[hh];       a0.h2[1] = ar[2 + hh];
  a1.h2[0] = ar[4 + hh];   a1.h2[1] = ar[6 + hh];
}
__device__ __forceinline__ void load_fragB(const _Float16* brow, int hh,
                                           fragAB& b0, fragAB& b1) {
  const v8h* br = (const v8h*)brow;
  b0.h2[0] = br[hh];       b0.h2[1] = br[2 + hh];
  b1.h2[0] = br[4 + hh];   b1.h2[1] = br[6 + hh];
}

// ---------------------------------------------------------------------------
// Fused q/k/v projection: three C[M,64] = A[M,64] @ W[64,64] + bias.
// A staged once (f32->f16), weights staged transposed, C staged through LDS
// for coalesced global_store_b128, no predication (M multiple of 32).
// ---------------------------------------------------------------------------
__global__ __launch_bounds__(256) void gemm64_qkv_k(
    const float* __restrict__ A,
    const float* __restrict__ W0, const float* __restrict__ bias0, float* __restrict__ C0,
    const float* __restrict__ W1, const float* __restrict__ bias1, float* __restrict__ C1,
    const float* __restrict__ W2, const float* __restrict__ bias2, float* __restrict__ C2) {
  __shared__ __align__(16) _Float16 aS[32 * 64];
  __shared__ __align__(16) _Float16 wT[3][64 * 64];
  __shared__ __align__(16) float    cS[32 * 64];
  const float* Ws[3]    = {W0, W1, W2};
  const float* biasV[3] = {bias0, bias1, bias2};
  float*       Cs[3]    = {C0, C1, C2};

  int tid = threadIdx.x;
  int row0 = blockIdx.x * 32;
  for (int i = tid; i < 32 * 64; i += 256)
    aS[i] = (_Float16)A[(size_t)row0 * 64 + i];
  for (int i = tid; i < 3 * 64 * 64; i += 256) {
    int mat = i >> 12, idx = i & 4095;
    int k = idx >> 6, n = idx & 63;
    wT[mat][n * 64 + k] = (_Float16)Ws[mat][idx];
  }
  __syncthreads();

  int wave = tid >> 5, lane = tid & 31;
  int hh = lane >> 4, mr = lane & 15;
  int m0 = (wave >> 2) * 16, n0 = (wave & 3) * 16;

  fragAB a0, a1;
  load_fragA(&aS[(m0 + mr) * 64], hh, a0, a1);

#pragma unroll
  for (int mat = 0; mat < 3; ++mat) {
    fragAB b0, b1;
    load_fragB(&wT[mat][(n0 + mr) * 64], hh, b0, b1);
    fragC c;
    float bv = biasV[mat][n0 + mr];
#pragma unroll
    for (int r = 0; r < 8; ++r) c.f[r] = bv;
    c.v = __builtin_amdgcn_wmma_f32_16x16x32_f16(false, a0.v, false, b0.v,
                                                 (short)0, c.v, false, false);
    c.v = __builtin_amdgcn_wmma_f32_16x16x32_f16(false, a1.v, false, b1.v,
                                                 (short)0, c.v, false, false);
#pragma unroll
    for (int r = 0; r < 8; ++r)
      cS[(m0 + r + 8 * hh) * 64 + n0 + mr] = c.f[r];
    __syncthreads();
    float4* Cg = (float4*)(Cs[mat] + (size_t)row0 * 64);
    const float4* cSv = (const float4*)cS;
#pragma unroll
    for (int i = tid; i < 512; i += 256) Cg[i] = cSv[i];
    __syncthreads();
  }
}

// ---------------------------------------------------------------------------
// Single-output GEMM with fused gelu (attention output projection).
// ---------------------------------------------------------------------------
__global__ __launch_bounds__(256) void gemm64_gelu_k(
    const float* __restrict__ A, const float* __restrict__ W,
    const float* __restrict__ bias, float* __restrict__ C) {
  __shared__ __align__(16) _Float16 aS[32 * 64];
  __shared__ __align__(16) _Float16 wT[64 * 64];
  __shared__ __align__(16) float    cS[32 * 64];
  int tid = threadIdx.x;
  int row0 = blockIdx.x * 32;
  for (int i = tid; i < 32 * 64; i += 256)
    aS[i] = (_Float16)A[(size_t)row0 * 64 + i];
  for (int i = tid; i < 64 * 64; i += 256) {
    int k = i >> 6, n = i & 63;
    wT[n * 64 + k] = (_Float16)W[i];
  }
  __syncthreads();

  int wave = tid >> 5, lane = tid & 31;
  int hh = lane >> 4, mr = lane & 15;
  int m0 = (wave >> 2) * 16, n0 = (wave & 3) * 16;

  fragAB a0, a1, b0, b1;
  load_fragA(&aS[(m0 + mr) * 64], hh, a0, a1);
  load_fragB(&wT[(n0 + mr) * 64], hh, b0, b1);
  fragC c;
  float bv = bias[n0 + mr];
#pragma unroll
  for (int r = 0; r < 8; ++r) c.f[r] = bv;
  c.v = __builtin_amdgcn_wmma_f32_16x16x32_f16(false, a0.v, false, b0.v,
                                               (short)0, c.v, false, false);
  c.v = __builtin_amdgcn_wmma_f32_16x16x32_f16(false, a1.v, false, b1.v,
                                               (short)0, c.v, false, false);
#pragma unroll
  for (int r = 0; r < 8; ++r)
    cS[(m0 + r + 8 * hh) * 64 + n0 + mr] = gelu_f(c.f[r]);
  __syncthreads();
  float4* Cg = (float4*)(C + (size_t)row0 * 64);
  const float4* cSv = (const float4*)cS;
#pragma unroll
  for (int i = tid; i < 512; i += 256) Cg[i] = cSv[i];
}

// ---------------------------------------------------------------------------
// Neighborhood vector attention. One wave per point; lane covers h and h+32.
// ---------------------------------------------------------------------------
__global__ __launch_bounds__(256) void neigh_attn_k(
    const float* __restrict__ q, const float* __restrict__ k,
    const float* __restrict__ v, const float* __restrict__ pe,
    const float* __restrict__ pts, const int* __restrict__ nidx,
    const float* __restrict__ pdw, const float* __restrict__ pdb,
    const float* __restrict__ aw, const float* __restrict__ ab,
    float* __restrict__ out) {
  int wave = threadIdx.x >> 5, lane = threadIdx.x & 31;
  int p = blockIdx.x * 8 + wave;
  int bs = p >> 10;
  int h0 = lane, h1 = lane + 32;
  float q0 = q[(size_t)p * 64 + h0] + pe[(size_t)p * 64 + h0];
  float q1 = q[(size_t)p * 64 + h1] + pe[(size_t)p * 64 + h1];
  float aw0 = aw[h0], aw1 = aw[h1];
  float px = pts[p * 3], py = pts[p * 3 + 1], pz = pts[p * 3 + 2];
  float lg[KNN], v0[KNN], v1[KNN];
#pragma unroll
  for (int kk = 0; kk < KNN; ++kk) {
    int m = nidx[p * KNN + kk];
    int g = (bs << 10) + m;
    float rx = pts[g * 3] - px, ry = pts[g * 3 + 1] - py, rz = pts[g * 3 + 2] - pz;
    float pd0 = pdb[h0] + rx * pdw[h0] + ry * pdw[64 + h0] + rz * pdw[128 + h0];
    float pd1 = pdb[h1] + rx * pdw[h1] + ry * pdw[64 + h1] + rz * pdw[128 + h1];
    float kk0 = k[(size_t)g * 64 + h0] + pe[(size_t)g * 64 + h0] + pd0;
    float kk1 = k[(size_t)g * 64 + h1] + pe[(size_t)g * 64 + h1] + pd1;
    float part = q0 * kk0 * aw0 + q1 * kk1 * aw1;
#pragma unroll
    for (int off = 16; off >= 1; off >>= 1) part += __shfl_xor(part, off);
    lg[kk] = part * 0.125f + ab[0];
    v0[kk] = v[(size_t)g * 64 + h0];
    v1[kk] = v[(size_t)g * 64 + h1];
  }
  float mx = lg[0];
#pragma unroll
  for (int kk = 1; kk < KNN; ++kk) mx = fmaxf(mx, lg[kk]);
  float sum = 0.f;
#pragma unroll
  for (int kk = 0; kk < KNN; ++kk) { lg[kk] = __expf(lg[kk] - mx); sum += lg[kk]; }
  float inv = 1.f / sum;
  float o0 = 0.f, o1 = 0.f;
#pragma unroll
  for (int kk = 0; kk < KNN; ++kk) { float w = lg[kk] * inv; o0 += w * v0[kk]; o1 += w * v1[kk]; }
  out[(size_t)p * 64 + h0] = o0;
  out[(size_t)p * 64 + h1] = o1;
}

// ---------------------------------------------------------------------------
// x = LayerNorm(x + a) * s + b  (wave per row, eps = 1e-6)
// ---------------------------------------------------------------------------
__global__ __launch_bounds__(256) void resid_ln_k(
    float* __restrict__ x, const float* __restrict__ a,
    const float* __restrict__ s, const float* __restrict__ b) {
  int wave = threadIdx.x >> 5, lane = threadIdx.x & 31;
  int p = blockIdx.x * 8 + wave;
  float v0 = x[(size_t)p * 64 + lane]      + a[(size_t)p * 64 + lane];
  float v1 = x[(size_t)p * 64 + lane + 32] + a[(size_t)p * 64 + lane + 32];
  float sum = v0 + v1;
#pragma unroll
  for (int off = 16; off >= 1; off >>= 1) sum += __shfl_xor(sum, off);
  float mean = sum * (1.f / 64.f);
  float d0 = v0 - mean, d1 = v1 - mean;
  float vs = d0 * d0 + d1 * d1;
#pragma unroll
  for (int off = 16; off >= 1; off >>= 1) vs += __shfl_xor(vs, off);
  float inv = rsqrtf(vs * (1.f / 64.f) + 1e-6f);
  x[(size_t)p * 64 + lane]      = d0 * inv * s[lane] + b[lane];
  x[(size_t)p * 64 + lane + 32] = d1 * inv * s[lane + 32] + b[lane + 32];
}

// pooled[bs,h] = max_n x[bs,n,h]
__global__ void rowmax_k(const float* __restrict__ x, float* __restrict__ pooled) {
  int i = blockIdx.x * blockDim.x + threadIdx.x;
  if (i >= 16 * HDIM) return;
  int bs = i >> 6, h = i & 63;
  const float* base = x + (size_t)bs * NPTS * 64 + h;
  float m = base[0];
  for (int n = 1; n < NPTS; ++n) m = fmaxf(m, base[(size_t)n * 64]);
  pooled[i] = m;
}

// ---------------------------------------------------------------------------
// Whole tail: encoder transformer over S=8 tokens, ln2, max over S, latent,
// dec_in, then the sequential 8-step decoder.  Single block.
// ---------------------------------------------------------------------------
__global__ __launch_bounds__(256) void enc_dec_tail_k(
    const float* __restrict__ pooled,
    const float* __restrict__ ekw, const float* __restrict__ ekb,
    const float* __restrict__ eqw, const float* __restrict__ eqb,
    const float* __restrict__ evw, const float* __restrict__ evb,
    const float* __restrict__ o1w, const float* __restrict__ o1b,
    const float* __restrict__ o2w, const float* __restrict__ o2b,
    const float* __restrict__ ln2s, const float* __restrict__ ln2b,
    const float* __restrict__ latw, const float* __restrict__ latb,
    const float* __restrict__ dinw, const float* __restrict__ dinb,
    const float* __restrict__ dqw, const float* __restrict__ dqb,
    const float* __restrict__ dkw, const float* __restrict__ dkb,
    const float* __restrict__ dvw, const float* __restrict__ dvb,
    const float* __restrict__ dow, const float* __restrict__ dob,
    float* __restrict__ xdec) {
  __shared__ float xin[16][64], qs[16][64], ks[16][64], vs[16][64];
  __shared__ float sc[2][8][8], att[16][64], h1[16][64], enc[16][64];
  __shared__ float mrow[16], vrow[16], p2[2][64], lat[2][32];
  __shared__ float decx[2][9][64], pe[9][64];
  __shared__ float kd[2][9][64], vd[2][9][64], qd[2][64], god[2][64], scd[2][9];
  int t = threadIdx.x;
  for (int i = t; i < 1024; i += 256) xin[i >> 6][i & 63] = pooled[i];
  __syncthreads();
  for (int i = t; i < 1024; i += 256) {
    int r = i >> 6, h = i & 63;
    float aq = eqb[h], ak = ekb[h], av = evb[h];
    for (int kk = 0; kk < 64; ++kk) {
      float xv = xin[r][kk];
      aq += xv * eqw[kk * 64 + h]; ak += xv * ekw[kk * 64 + h]; av += xv * evw[kk * 64 + h];
    }
    qs[r][h] = aq; ks[r][h] = ak; vs[r][h] = av;
  }
  __syncthreads();
  if (t < 128) {
    int b = t >> 6, r = (t >> 3) & 7, s = t & 7;
    float acc = 0.f;
    for (int h = 0; h < 64; ++h) acc += qs[b * 8 + r][h] * ks[b * 8 + s][h];
    sc[b][r][s] = acc * 0.125f;
  }
  __syncthreads();
  if (t < 16) {
    int b = t >> 3, r = t & 7;
    float mx = sc[b][r][0];
    for (int s = 1; s < 8; ++s) mx = fmaxf(mx, sc[b][r][s]);
    float sum = 0.f, e[8];
    for (int s = 0; s < 8; ++s) { e[s] = __expf(sc[b][r][s] - mx); sum += e[s]; }
    for (int s = 0; s < 8; ++s) sc[b][r][s] = e[s] / sum;
  }
  __syncthreads();
  for (int i = t; i < 1024; i += 256) {
    int r = i >> 6, h = i & 63, b = r >> 3, s0 = r & 7;
    float acc = 0.f;
    for (int s = 0; s < 8; ++s) acc += sc[b][s0][s] * vs[b * 8 + s][h];
    att[r][h] = acc;
  }
  __syncthreads();
  for (int i = t; i < 1024; i += 256) {
    int r = i >> 6, h = i & 63;
    float acc = o1b[h];
    for (int kk = 0; kk < 64; ++kk) acc += att[r][kk] * o1w[kk * 64 + h];
    h1[r][h] = gelu_f(acc);
  }
  __syncthreads();
  for (int i = t; i < 1024; i += 256) {
    int r = i >> 6, h = i & 63;
    float acc = o2b[h];
    for (int kk = 0; kk < 64; ++kk) acc += h1[r][kk] * o2w[kk * 64 + h];
    enc[r][h] = xin[r][h] + acc;
  }
  __syncthreads();
  if (t < 16) {
    float m = 0.f;
    for (int h = 0; h < 64; ++h) m += enc[t][h];
    m *= (1.f / 64.f);
    float v = 0.f;
    for (int h = 0; h < 64; ++h) { float d = enc[t][h] - m; v += d * d; }
    mrow[t] = m; vrow[t] = rsqrtf(v * (1.f / 64.f) + 1e-6f);
  }
  __syncthreads();
  for (int i = t; i < 1024; i += 256) {
    int r = i >> 6, h = i & 63;
    enc[r][h] = (enc[r][h] - mrow[r]) * vrow[r] * ln2s[h] + ln2b[h];
  }
  __syncthreads();
  if (t < 128) {
    int b = t >> 6, h = t & 63;
    float mx = enc[b * 8][h];
    for (int s = 1; s < 8; ++s) mx = fmaxf(mx, enc[b * 8 + s][h]);
    p2[b][h] = mx;
  }
  __syncthreads();
  if (t < 64) {
    int b = t >> 5, l = t & 31;
    float acc = latb[l];
    for (int h = 0; h < 64; ++h) acc += p2[b][h] * latw[h * 32 + l];
    lat[b][l] = acc;
  }
  __syncthreads();
  if (t < 128) {
    int b = t >> 6, h = t & 63;
    float acc = dinb[h];
    for (int l = 0; l < 32; ++l) acc += lat[b][l] * dinw[l * 64 + h];
    decx[b][0][h] = gelu_f(acc);
  }
  for (int i = t; i < 576; i += 256) {
    int p = i >> 6, h = i & 63;
    float d = __expf(-(float)(h & ~1) * (9.21034037197618f / 64.f));  // ln(10000)/H
    float ph = (float)p * d;
    pe[p][h] = (h & 1) ? __cosf(ph) : __sinf(ph);
  }
  __syncthreads();
  for (int it = 0; it < 8; ++it) {
    int tcnt = it + 1;
    const float* qw  = dqw + it * 4096; const float* qbv = dqb + it * 64;
    const float* kw  = dkw + it * 4096; const float* kbv = dkb + it * 64;
    const float* vw  = dvw + it * 4096; const float* vbv = dvb + it * 64;
    const float* ow  = dow + it * 4096; const float* obv = dob + it * 64;
    for (int i = t; i < 2 * tcnt * 64; i += 256) {
      int b = i / (tcnt * 64); int rem = i - b * tcnt * 64;
      int j = rem >> 6, h = rem & 63;
      float ak = kbv[h], av = vbv[h];
      for (int kk = 0; kk < 64; ++kk) {
        float xv = decx[b][j][kk] + pe[j][kk];
        ak += xv * kw[kk * 64 + h]; av += xv * vw[kk * 64 + h];
      }
      kd[b][j][h] = ak; vd[b][j][h] = av;
    }
    if (t < 128) {
      int b = t >> 6, h = t & 63;
      float aq = qbv[h];
      for (int kk = 0; kk < 64; ++kk) {
        float xv = decx[b][tcnt - 1][kk] + pe[tcnt - 1][kk];
        aq += xv * qw[kk * 64 + h];
      }
      qd[b][h] = aq;
    }
    __syncthreads();
    if (t < 2 * tcnt) {
      int b = t / tcnt, j = t - b * tcnt;
      float acc = 0.f;
      for (int h = 0; h < 64; ++h) acc += qd[b][h] * kd[b][j][h];
      scd[b][j] = acc * 0.125f;
    }
    __syncthreads();
    if (t < 2) {
      float mx = scd[t][0];
      for (int j = 1; j < tcnt; ++j) mx = fmaxf(mx, scd[t][j]);
      float sum = 0.f;
      for (int j = 0; j < tcnt; ++j) { float e = __expf(scd[t][j] - mx); scd[t][j] = e; sum += e; }
      float inv = 1.f / sum;
      for (int j = 0; j < tcnt; ++j) scd[t][j] *= inv;
    }
    __syncthreads();
    if (t < 128) {
      int b = t >> 6, h = t & 63;
      float acc = 0.f;
      for (int j = 0; j < tcnt; ++j) acc += scd[b][j] * vd[b][j][h];
      god[b][h] = gelu_f(acc);
    }
    __syncthreads();
    if (t < 128) {
      int b = t >> 6, h = t & 63;
      float acc = obv[h];
      for (int kk = 0; kk < 64; ++kk) acc += god[b][kk] * ow[kk * 64 + h];
      decx[b][tcnt][h] = decx[b][0][h] + acc;
    }
    __syncthreads();
  }
  for (int i = t; i < 1024; i += 256) {
    int r = i >> 6, h = i & 63, b = r >> 3, s = r & 7;
    xdec[i] = decx[b][s + 1][h];
  }
}

// coarse[bs, j, d] = xdec[bs] @ coarse_w + coarse_b  (16 x 768 outputs)
__global__ __launch_bounds__(256) void coarse_k(
    const float* __restrict__ xdec, const float* __restrict__ W,
    const float* __restrict__ b, float* __restrict__ out) {
  int i = blockIdx.x * blockDim.x + threadIdx.x;
  if (i >= 16 * 768) return;
  int bs = i / 768, c = i % 768;
  float acc = b[c];
  const float* xr = xdec + bs * 64;
  for (int h = 0; h < 64; ++h) acc += xr[h] * W[h * 768 + c];
  out[i] = acc;
}

// feat = gelu(LN(coarse@feat_w+feat_b)); off = feat@off_w+off_b; fine = coarse+off
__global__ __launch_bounds__(256) void fine_k(
    const float* __restrict__ coarse,
    const float* __restrict__ fw, const float* __restrict__ fb,
    const float* __restrict__ lns, const float* __restrict__ lnb,
    const float* __restrict__ ow, const float* __restrict__ ob,
    float* __restrict__ out) {
  int i = blockIdx.x * blockDim.x + threadIdx.x;
  if (i >= 4096) return;
  int bs = i >> 8, j = i & 255;
  float c0 = coarse[i * 3], c1 = coarse[i * 3 + 1], c2 = coarse[i * 3 + 2];
  float f[64];
  float mean = 0.f;
#pragma unroll
  for (int h = 0; h < 64; ++h) {
    f[h] = fb[h] + c0 * fw[h] + c1 * fw[64 + h] + c2 * fw[128 + h];
    mean += f[h];
  }
  mean *= (1.f / 64.f);
  float var = 0.f;
#pragma unroll
  for (int h = 0; h < 64; ++h) { float d = f[h] - mean; var += d * d; }
  float inv = rsqrtf(var * (1.f / 64.f) + 1e-6f);
#pragma unroll
  for (int h = 0; h < 64; ++h) f[h] = gelu_f((f[h] - mean) * inv * lns[h] + lnb[h]);
#pragma unroll
  for (int u = 0; u < 4; ++u)
#pragma unroll
    for (int d = 0; d < 3; ++d) {
      float acc = ob[u * 3 + d];
      for (int h = 0; h < 64; ++h) acc += f[h] * ow[h * 12 + u * 3 + d];
      float cd = (d == 0) ? c0 : ((d == 1) ? c1 : c2);
      out[((size_t)(bs * 1024 + j * 4 + u)) * 3 + d] = cd + acc;
    }
}

// ---------------------------------------------------------------------------
// Input layout (setup_inputs dict order, nested leaves flattened):
// 0 points; 1 enc_in_w; 2 enc_in_b;
// 3..50 ptl[3] x {qw,qb,kw,kb,vw,vb,pew,peb,pdw,pdb,aw,ab,ow,ob,lns,lnb}
// 51..60 etl {kw,kb,qw,qb,vw,vb,o1w,o1b,o2w,o2b}
// 61 ln2_s; 62 ln2_b; 63 lat_w; 64 lat_b; 65 dec_in_w; 66 dec_in_b;
// 67..74 dec {qw,qb,kw,kb,vw,vb,ow,ob}; 75 coarse_w; 76 coarse_b;
// 77 feat_w; 78 feat_b; 79 feat_ln_s; 80 feat_ln_b; 81 off_w; 82 off_b
// ---------------------------------------------------------------------------
extern "C" void kernel_launch(void* const* d_in, const int* in_sizes, int n_in,
                              void* d_out, int out_size, void* d_ws, size_t ws_size,
                              hipStream_t stream) {
  (void)in_sizes; (void)n_in; (void)out_size; (void)ws_size;
  const float* pts = (const float*)d_in[0];
  auto P = [&](int i) { return (const float*)d_in[i]; };

  float* ws     = (float*)d_ws;
  const size_t BUF = (size_t)BSN * HDIM;  // 1,048,576 floats
  float* x      = ws;
  float* qb     = ws + 1 * BUF;
  float* kb     = ws + 2 * BUF;
  float* vb     = ws + 3 * BUF;
  float* pe     = ws + 4 * BUF;
  float* ao     = ws + 5 * BUF;
  int*   nidx   = (int*)(ws + 6 * BUF);          // 262,144 ints
  float* pooled = ws + 6 * BUF + 262144;         // 1024
  float* xdec   = pooled + 1024;                 // 1024
  float* coarse = xdec + 1024;                   // 12288

  dense3_k<<<4096, 256, 0, stream>>>(pts, P(1), P(2), x, BSN);
  knn_k<<<64, 256, 0, stream>>>(pts, nidx);

  for (int L = 0; L < 3; ++L) {
    int base = 3 + L * 16;
    gemm64_qkv_k<<<512, 256, 0, stream>>>(x,
        P(base + 0), P(base + 1), qb,
        P(base + 2), P(base + 3), kb,
        P(base + 4), P(base + 5), vb);
    dense3_k<<<4096, 256, 0, stream>>>(pts, P(base + 6), P(base + 7), pe, BSN);
    neigh_attn_k<<<2048, 256, 0, stream>>>(qb, kb, vb, pe, pts, nidx,
                                           P(base + 8), P(base + 9),
                                           P(base + 10), P(base + 11), ao);
    gemm64_gelu_k<<<512, 256, 0, stream>>>(ao, P(base + 12), P(base + 13), qb);
    resid_ln_k<<<2048, 256, 0, stream>>>(x, qb, P(base + 14), P(base + 15));
  }

  rowmax_k<<<4, 256, 0, stream>>>(x, pooled);
  enc_dec_tail_k<<<1, 256, 0, stream>>>(pooled,
      P(51), P(52), P(53), P(54), P(55), P(56), P(57), P(58), P(59), P(60),
      P(61), P(62), P(63), P(64), P(65), P(66),
      P(67), P(68), P(69), P(70), P(71), P(72), P(73), P(74),
      xdec);
  coarse_k<<<48, 256, 0, stream>>>(xdec, P(75), P(76), coarse);
  fine_k<<<16, 256, 0, stream>>>(coarse, P(77), P(78), P(79), P(80), P(81), P(82),
                                 (float*)d_out);
}
